// LinearMessagePassingLayer_62749472195030
// MI455X (gfx1250) — compile-verified, matrix operand order checked by reading, not run on hardware
//
#include <hip/hip_runtime.h>
#include <hip/hip_bf16.h>
#include <cstdint>
#include <cstddef>

// ---------------------------------------------------------------------------
// Types for CDNA5 WMMA (wave32): v_wmma_f32_16x16x32_bf16
// ---------------------------------------------------------------------------
typedef __attribute__((ext_vector_type(16))) __bf16          v16bf;
typedef __attribute__((ext_vector_type(8)))  float           v8f;
typedef __attribute__((ext_vector_type(4)))  float           v4f;

#define NNODES 50000
#define NEDGES 600000
#define DIM    128
#define LN_EPS 1e-6f

__device__ __forceinline__ unsigned short f2bf(float f) {
  unsigned u = __float_as_uint(f);
  unsigned r = u + 0x7FFFu + ((u >> 16) & 1u);   // round-to-nearest-even
  return (unsigned short)(r >> 16);
}

__device__ __forceinline__ v8f wmma_bf16(v16bf a, v16bf b, v8f c) {
  return __builtin_amdgcn_wmma_f32_16x16x32_bf16(
      /*neg_a=*/false, a, /*neg_b=*/false, b,
      /*c_mod=*/(short)0, c, /*reuse_a=*/false, /*reuse_b=*/false);
}

// Opaque 32-bit zero in a VGPR: defeats LICM/CSE of loads whose address adds
// it, WITHOUT laundering the base pointer (keeps global addrspace inference).
__device__ __forceinline__ unsigned opaque_zero() {
  unsigned k = 0;
  asm volatile("" : "+v"(k));
  return k;
}

// Per-lane K index for the 16-bit 16x32 A (and mirrored B) fragment layout:
//   K(h, lane) = h + (h & 8) + ((lane & 16) >> 1)

// ---------------------------------------------------------------------------
// Pack a (KT*32 x 128) f32 weight matrix into bf16 WMMA B fragments.
// frag = blockIdx.x = kt*8 + nt ; 32 threads = 1 wave, 16 bf16 per lane.
// Fragment storage (bytes): frag*1024 + lane*32, 32 bytes per lane.
// ---------------------------------------------------------------------------
__global__ void pack_weight_kernel(const float* __restrict__ W,
                                   unsigned short* __restrict__ out) {
  const int frag = blockIdx.x;
  const int kt = frag >> 3, nt = frag & 7;
  const int lane = threadIdx.x & 31;
  const int col = nt * 16 + (lane & 15);
  const int koff = (lane & 16) >> 1;
  unsigned short* o = out + (unsigned)((frag * 32 + lane) * 16);
#pragma unroll
  for (int h = 0; h < 16; ++h) {
    const int k = kt * 32 + h + (h & 8) + koff;
    o[h] = f2bf(W[(unsigned)(k * DIM + col)]);
  }
}

__device__ __forceinline__ v16bf load_a_frag(const float* base, unsigned off) {
  // base+off points at the lane's low 8-float run; high run is 16 floats on.
  const float* s = base + off;
  v4f x0 = *(const v4f*)(s);
  v4f x1 = *(const v4f*)(s + 4);
  v4f y0 = *(const v4f*)(s + 16);
  v4f y1 = *(const v4f*)(s + 20);
  v16bf b;
#pragma unroll
  for (int j = 0; j < 4; ++j) {
    b[j]      = (__bf16)x0[j];
    b[j + 4]  = (__bf16)x1[j];
    b[j + 8]  = (__bf16)y0[j];
    b[j + 12] = (__bf16)y1[j];
  }
  return b;
}

// vbyte = knock + nt*1024 + lane*32 (one VGPR offset per nt iteration);
// the kt stride (8 KB) is uniform -> folds into saddr/immediate.
__device__ __forceinline__ v16bf load_b_frag(const unsigned short* pack,
                                             unsigned vbyte, int kt) {
  const char* p = (const char*)pack + (unsigned)(kt * 8192);
  return *(const v16bf*)(p + vbyte);
}

// ---------------------------------------------------------------------------
// Edge kernel: per 16-edge tile, A = [nodes[senders] | edges] (16x256),
// B = W_message (256x128), scatter-add result into aggregated[receivers].
// ---------------------------------------------------------------------------
__global__ void __launch_bounds__(256)
edge_message_kernel(const float* __restrict__ nodes,
                    const float* __restrict__ edges,
                    const int* __restrict__ senders,
                    const int* __restrict__ receivers,
                    const unsigned short* __restrict__ packWm,
                    float* __restrict__ agg) {
  const int lane   = threadIdx.x & 31;
  const int wave   = (int)((blockIdx.x * blockDim.x + threadIdx.x) >> 5);
  const int nWaves = (int)((gridDim.x * blockDim.x) >> 5);
  const int koff   = (lane & 16) >> 1;
  const int row    = lane & 15;
  const unsigned lanebyte = (unsigned)(lane * 32);

  for (int t = wave; t < NEDGES / 16; t += nWaves) {
    const int e0 = t * 16;
    const int e  = e0 + row;
    const unsigned nbase = (unsigned)(senders[e] * DIM);   // gather row
    const unsigned ebase = (unsigned)(e * DIM);

    v16bf a[8];
#pragma unroll
    for (int kt = 0; kt < 8; ++kt) {
      const int c0 = kt * 32 + koff;
      if (kt < 4) a[kt] = load_a_frag(nodes, nbase + (unsigned)c0);
      else        a[kt] = load_a_frag(edges, ebase + (unsigned)(c0 - DIM));
    }

    int recs[8];
#pragma unroll
    for (int r = 0; r < 8; ++r) recs[r] = receivers[e0 + r + koff];

#pragma unroll 1   // keep B-fragment live range to one nt tile (no hoisting)
    for (int nt = 0; nt < 8; ++nt) {
      const unsigned vbyte = opaque_zero() + (unsigned)(nt * 1024) + lanebyte;
      v8f acc = {};
#pragma unroll
      for (int kt = 0; kt < 8; ++kt)
        acc = wmma_bf16(a[kt], load_b_frag(packWm, vbyte, kt), acc);
      const int c = nt * 16 + (lane & 15);
#pragma unroll
      for (int r = 0; r < 8; ++r)
        atomicAdd(agg + (unsigned)(recs[r] * DIM + c), acc[r]);
    }
  }
}

// ---------------------------------------------------------------------------
// Node kernel: fused 3-layer MLP + residual GEMM + LayerNorm per 16-node tile.
// ---------------------------------------------------------------------------
__global__ void __launch_bounds__(256)
node_update_kernel(const float* __restrict__ nodes,
                   const float* __restrict__ agg,
                   const unsigned short* __restrict__ pW0,
                   const unsigned short* __restrict__ pW1,
                   const unsigned short* __restrict__ pW2,
                   const unsigned short* __restrict__ pWn,
                   const float* __restrict__ b0,
                   const float* __restrict__ b1,
                   const float* __restrict__ b2,
                   const float* __restrict__ ln_scale,
                   const float* __restrict__ ln_bias,
                   float* __restrict__ out) {
  __shared__ __align__(16) float hbuf[8][16 * DIM];   // 64 KB, per-wave 8 KB
  const int lane   = threadIdx.x & 31;
  const int wl     = threadIdx.x >> 5;
  const int wave   = (int)((blockIdx.x * blockDim.x + threadIdx.x) >> 5);
  const int nWaves = (int)((gridDim.x * blockDim.x) >> 5);
  const int koff   = (lane & 16) >> 1;
  const int row    = lane & 15;
  const int col    = lane & 15;
  const unsigned lanebyte = (unsigned)(lane * 32);
  float* h = hbuf[wl];

  for (int t = wave; t < NNODES / 16; t += nWaves) {
    const int n0 = t * 16;
    const unsigned rbase = (unsigned)((n0 + row) * DIM);

    // A fragments for layer 0: cols 0..127 = nodes, 128..255 = aggregated
    v16bf a0[8];
#pragma unroll
    for (int kt = 0; kt < 8; ++kt) {
      const int c0 = kt * 32 + koff;
      if (kt < 4) a0[kt] = load_a_frag(nodes, rbase + (unsigned)c0);
      else        a0[kt] = load_a_frag(agg,   rbase + (unsigned)(c0 - DIM));
    }

    // ---- layer 0: relu(concat @ W0 + b0) -> LDS ----
#pragma unroll 1
    for (int nt = 0; nt < 8; ++nt) {
      const unsigned vbyte = opaque_zero() + (unsigned)(nt * 1024) + lanebyte;
      v8f acc = {};
#pragma unroll
      for (int kt = 0; kt < 8; ++kt)
        acc = wmma_bf16(a0[kt], load_b_frag(pW0, vbyte, kt), acc);
      const int c = nt * 16 + col;
      const float bb = b0[c];
#pragma unroll
      for (int r = 0; r < 8; ++r)
        h[(r + koff) * DIM + c] = fmaxf(acc[r] + bb, 0.0f);
    }
    asm volatile("s_wait_dscnt 0x0" ::: "memory");
    __builtin_amdgcn_wave_barrier();

    // relayout h -> A fragments (K = 128)
    v16bf a1[4];
#pragma unroll
    for (int kt = 0; kt < 4; ++kt)
      a1[kt] = load_a_frag(h, (unsigned)(row * DIM + kt * 32 + koff));
    asm volatile("s_wait_dscnt 0x0" ::: "memory");
    __builtin_amdgcn_wave_barrier();

    // ---- layer 1: relu(h0 @ W1 + b1) -> LDS ----
#pragma unroll 1
    for (int nt = 0; nt < 8; ++nt) {
      const unsigned vbyte = opaque_zero() + (unsigned)(nt * 1024) + lanebyte;
      v8f acc = {};
#pragma unroll
      for (int kt = 0; kt < 4; ++kt)
        acc = wmma_bf16(a1[kt], load_b_frag(pW1, vbyte, kt), acc);
      const int c = nt * 16 + col;
      const float bb = b1[c];
#pragma unroll
      for (int r = 0; r < 8; ++r)
        h[(r + koff) * DIM + c] = fmaxf(acc[r] + bb, 0.0f);
    }
    asm volatile("s_wait_dscnt 0x0" ::: "memory");
    __builtin_amdgcn_wave_barrier();

    v16bf a2[4];
#pragma unroll
    for (int kt = 0; kt < 4; ++kt)
      a2[kt] = load_a_frag(h, (unsigned)(row * DIM + kt * 32 + koff));
    asm volatile("s_wait_dscnt 0x0" ::: "memory");
    __builtin_amdgcn_wave_barrier();

    // ---- layer 2 + residual: y = h1 @ W2 + b2 + nodes @ W_node -> LDS ----
#pragma unroll 1
    for (int nt = 0; nt < 8; ++nt) {
      const unsigned vbyte = opaque_zero() + (unsigned)(nt * 1024) + lanebyte;
      v8f acc = {};
#pragma unroll
      for (int kt = 0; kt < 4; ++kt)
        acc = wmma_bf16(a2[kt], load_b_frag(pW2, vbyte, kt), acc);
#pragma unroll
      for (int kt = 0; kt < 4; ++kt)
        acc = wmma_bf16(a0[kt], load_b_frag(pWn, vbyte, kt), acc);
      const int c = nt * 16 + col;
      const float bb = b2[c];
#pragma unroll
      for (int r = 0; r < 8; ++r)
        h[(r + koff) * DIM + c] = acc[r] + bb;
    }
    asm volatile("s_wait_dscnt 0x0" ::: "memory");
    __builtin_amdgcn_wave_barrier();

    // ---- LayerNorm from LDS: each lane owns half a row (64 cols) ----
    {
      const int lrow = lane >> 1;           // 0..15
      const int half = (lane & 1) * 64;     // 0 or 64
      const float* hr = h + lrow * DIM + half;
      float s = 0.f, s2 = 0.f;
#pragma unroll
      for (int j = 0; j < 16; ++j) {
        const v4f v = *(const v4f*)(hr + j * 4);
#pragma unroll
        for (int cidx = 0; cidx < 4; ++cidx) { s += v[cidx]; s2 += v[cidx] * v[cidx]; }
      }
      s  += __shfl_xor(s,  1);
      s2 += __shfl_xor(s2, 1);
      const float mu  = s * (1.0f / DIM);
      const float var = s2 * (1.0f / DIM) - mu * mu;
      const float rs  = rsqrtf(var + LN_EPS);

      float* orow = out + (unsigned)((n0 + lrow) * DIM + half);
#pragma unroll
      for (int j = 0; j < 16; ++j) {
        const v4f v = *(const v4f*)(hr + j * 4);
        const v4f g = *(const v4f*)(ln_scale + half + j * 4);
        const v4f be = *(const v4f*)(ln_bias + half + j * 4);
        v4f w;
#pragma unroll
        for (int cidx = 0; cidx < 4; ++cidx)
          w[cidx] = (v[cidx] - mu) * rs * g[cidx] + be[cidx];
        *(v4f*)(orow + j * 4) = w;
      }
    }
    asm volatile("s_wait_dscnt 0x0" ::: "memory");  // protect h reuse next tile
    __builtin_amdgcn_wave_barrier();
  }
}

// ---------------------------------------------------------------------------
// Host-side launch
// ---------------------------------------------------------------------------
extern "C" void kernel_launch(void* const* d_in, const int* in_sizes, int n_in,
                              void* d_out, int out_size, void* d_ws, size_t ws_size,
                              hipStream_t stream) {
  const float* nodes     = (const float*)d_in[0];
  const float* edges     = (const float*)d_in[1];
  const int*   senders   = (const int*)d_in[2];
  const int*   receivers = (const int*)d_in[3];
  const float* W_message = (const float*)d_in[4];
  const float* W_node    = (const float*)d_in[5];
  const float* mlp_w0    = (const float*)d_in[6];
  const float* mlp_b0    = (const float*)d_in[7];
  const float* mlp_w1    = (const float*)d_in[8];
  const float* mlp_b1    = (const float*)d_in[9];
  const float* mlp_w2    = (const float*)d_in[10];
  const float* mlp_b2    = (const float*)d_in[11];
  const float* ln_scale  = (const float*)d_in[12];
  const float* ln_bias   = (const float*)d_in[13];
  float* out = (float*)d_out;

  char* ws = (char*)d_ws;
  float* agg = (float*)ws;
  size_t off = (size_t)NNODES * DIM * sizeof(float);         // 25.6 MB
  unsigned short* pWm = (unsigned short*)(ws + off); off += 64 * 32 * 16 * 2; // 64 KB
  unsigned short* pW0 = (unsigned short*)(ws + off); off += 64 * 32 * 16 * 2; // 64 KB
  unsigned short* pW1 = (unsigned short*)(ws + off); off += 32 * 32 * 16 * 2; // 32 KB
  unsigned short* pW2 = (unsigned short*)(ws + off); off += 32 * 32 * 16 * 2; // 32 KB
  unsigned short* pWn = (unsigned short*)(ws + off); off += 32 * 32 * 16 * 2; // 32 KB

  // Zero the aggregation buffer (capture-legal async memset).
  hipMemsetAsync(agg, 0, (size_t)NNODES * DIM * sizeof(float), stream);

  // Pack weights into WMMA B-fragment layout (bf16).
  pack_weight_kernel<<<64, 32, 0, stream>>>(W_message, pWm);
  pack_weight_kernel<<<64, 32, 0, stream>>>(mlp_w0,    pW0);
  pack_weight_kernel<<<32, 32, 0, stream>>>(mlp_w1,    pW1);
  pack_weight_kernel<<<32, 32, 0, stream>>>(mlp_w2,    pW2);
  pack_weight_kernel<<<32, 32, 0, stream>>>(W_node,    pWn);

  // Edge GEMM + scatter: 37500 tiles, 8 waves / 256-thread block.
  edge_message_kernel<<<(NEDGES / 16 + 7) / 8, 256, 0, stream>>>(
      nodes, edges, senders, receivers, pWm, agg);

  // Fused node MLP + residual + LayerNorm: 3125 tiles.
  node_update_kernel<<<(NNODES / 16 + 7) / 8, 256, 0, stream>>>(
      nodes, agg, pW0, pW1, pW2, pWn,
      mlp_b0, mlp_b1, mlp_b2, ln_scale, ln_bias, out);

  (void)in_sizes; (void)n_in; (void)out_size; (void)ws_size;
}
